// CondConvBinaryClassifier_87359634801419
// MI455X (gfx1250) — matrix operand order repaired and checked by reading, not compiled, not run on hardware
//
#include <hip/hip_runtime.h>
#include <hip/hip_bf16.h>

typedef __attribute__((ext_vector_type(16))) _Float16 v16h;
typedef __attribute__((ext_vector_type(8)))  float    v8f;
typedef __attribute__((ext_vector_type(4)))  unsigned int u32x4;
typedef __attribute__((ext_vector_type(8)))  int      i32x8;
typedef __attribute__((ext_vector_type(4)))  int      i32x4;

#define BN_EPS 1e-5f

// ---------------------------------------------------------------------------
// Stem: conv 3x3 s2 p1 (3->32) + BN + ReLU, fp32 in -> f16 out [64][32][112*112]
// ---------------------------------------------------------------------------
__global__ void stem_kernel(const float* __restrict__ x, const float* __restrict__ w,
                            const float* __restrict__ g, const float* __restrict__ be,
                            const float* __restrict__ m, const float* __restrict__ v,
                            _Float16* __restrict__ out)
{
    const int S = 112 * 112;
    int idx = blockIdx.x * blockDim.x + threadIdx.x;          // [64][32][S]
    if (idx >= 64 * 32 * S) return;
    int p = idx % S; int t = idx / S; int c = t % 32; int b = t / 32;
    int y = p / 112, xo = p % 112;
    float acc = 0.f;
    for (int ci = 0; ci < 3; ++ci)
        for (int ky = 0; ky < 3; ++ky) {
            int hi = 2 * y - 1 + ky;
            if (hi < 0 || hi >= 224) continue;
            for (int kx = 0; kx < 3; ++kx) {
                int wi = 2 * xo - 1 + kx;
                if (wi < 0 || wi >= 224) continue;
                acc += x[((size_t)(b * 3 + ci) * 224 + hi) * 224 + wi] *
                       w[((c * 3 + ci) * 3 + ky) * 3 + kx];
            }
        }
    float sc  = g[c] * rsqrtf(v[c] + BN_EPS);
    float val = acc * sc + (be[c] - m[c] * sc);
    out[idx] = (_Float16)fmaxf(val, 0.f);
}

// ---------------------------------------------------------------------------
// Global average pool over spatial dim: f16 [B*C][S] -> f32 [B*C]
// ---------------------------------------------------------------------------
template<int S>
__global__ void gap_kernel(const _Float16* __restrict__ h, float* __restrict__ pooled)
{
    __shared__ float red[256];
    int bc = blockIdx.x;
    const _Float16* base = h + (size_t)bc * S;
    float s = 0.f;
    for (int i = threadIdx.x; i < S; i += 256) s += (float)base[i];
    red[threadIdx.x] = s;
    __syncthreads();
    for (int st = 128; st > 0; st >>= 1) {
        if (threadIdx.x < st) red[threadIdx.x] += red[threadIdx.x + st];
        __syncthreads();
    }
    if (threadIdx.x == 0) pooled[bc] = red[0] / (float)S;
}

// ---------------------------------------------------------------------------
// Routing: r[b][k] = sigmoid(pooled[b] . rw[k] + rb[k]); 64 batches x 4 experts
// ---------------------------------------------------------------------------
template<int C>
__global__ void route_kernel(const float* __restrict__ pooled, const float* __restrict__ rw,
                             const float* __restrict__ rb, float* __restrict__ r)
{
    int t = threadIdx.x;            // 256 = 64*4
    int b = t >> 2, k = t & 3;
    float z = rb[k];
    for (int c = 0; c < C; ++c) z += pooled[b * C + c] * rw[k * C + c];
    r[b * 4 + k] = 1.f / (1.f + __expf(-z));
}

// ---------------------------------------------------------------------------
// Mix expert kernels per-sample, pack into WMMA A-fragment layout (f16).
// K dimension is TAP-MAJOR: K = tap*CIN + c  (c = K & (CIN-1), tap = K >> lg2CIN)
// A 16x32 f16 tile lane layout: lane<16 -> K {0..15},
// lane>=16 -> K {16..31}; row M = lane%16. One v16h (32B) per lane.
// pack[((b*nMt + mtile)*nKb + kblk)*512 + lane*16 + h]
// ---------------------------------------------------------------------------
template<int CIN, int COUT>
__global__ void mixw_kernel(const float* __restrict__ r, const float* __restrict__ w,
                            _Float16* __restrict__ pack)
{
    const int nKb = CIN * 9 / 32, nMt = COUT / 16;
    const int CSH = (CIN == 32) ? 5 : 6;
    int tid = blockIdx.x * blockDim.x + threadIdx.x;   // 64*nMt*nKb*32 threads
    if (tid >= 64 * nMt * nKb * 32) return;
    int lane  = tid & 31;
    int kblk  = (tid >> 5) % nKb;
    int mtile = (tid / (32 * nKb)) % nMt;
    int b     = tid / (32 * nKb * nMt);
    int row   = lane & 15;
    int kb    = (lane < 16) ? 0 : 16;
    int cout  = mtile * 16 + row;
    float r0 = r[b * 4 + 0], r1 = r[b * 4 + 1], r2 = r[b * 4 + 2], r3 = r[b * 4 + 3];
    const size_t es = (size_t)COUT * CIN * 9;          // expert stride
    _Float16* dst = pack + (size_t)tid * 16;
    for (int h = 0; h < 16; ++h) {
        int koff = kb + h;
        int K = kblk * 32 + koff;
        int c = K & (CIN - 1), t9 = K >> CSH;          // tap-major
        size_t wi = ((size_t)cout * CIN + c) * 9 + t9; // w[e][cout][c][t9]
        float val = r0 * w[wi] + r1 * w[wi + es] + r2 * w[wi + 2 * es] + r3 * w[wi + 3 * es];
        dst[h] = (_Float16)val;
    }
}

// Mixed per-sample bias: cb[b][c] = sum_k r[b][k] * bias[k][c]
template<int COUT>
__global__ void mixb_kernel(const float* __restrict__ r, const float* __restrict__ bias,
                            float* __restrict__ cb)
{
    int tid = blockIdx.x * blockDim.x + threadIdx.x;
    if (tid >= 64 * COUT) return;
    int b = tid / COUT, c = tid % COUT;
    cb[tid] = r[b * 4 + 0] * bias[c] + r[b * 4 + 1] * bias[COUT + c] +
              r[b * 4 + 2] * bias[2 * COUT + c] + r[b * 4 + 3] * bias[3 * COUT + c];
}

// ---------------------------------------------------------------------------
// CondConv as WMMA implicit GEMM, per-image D[Cout x N] = W[Cout x K] * im2col.
// - A (per-sample packed weights, MT*nKb*512 f16 = 36KB) DMA'd into LDS once
//   per block by wave 0 via TENSOR_LOAD_TO_LDS (TDM), published with
//   s_wait_tensorcnt + workgroup barrier.
// - B im2col tile [NBLK x 32] staged in LDS per 32-wide K step (tap-major K).
// - 8 waves: wave -> (mtl = w%MT, nsub = w/MT); each wave holds NACC
//   accumulators (adjacent 16-wide N subtiles) sharing one A fragment.
// - Fused epilogue: + mixed bias, BN, ReLU, f16 store.
// ---------------------------------------------------------------------------
template<int CIN, int COUT, int MT, int NACC, int STRIDE, int HIN, int WIN, int HOUT, int WOUT>
__global__ __launch_bounds__(256)
void condconv_wmma(const _Float16* __restrict__ hin, const _Float16* __restrict__ wpack,
                   const float* __restrict__ cb,
                   const float* __restrict__ g, const float* __restrict__ be,
                   const float* __restrict__ m, const float* __restrict__ vv,
                   _Float16* __restrict__ hout)
{
    const int nKb   = CIN * 9 / 32;       // 288->9, 576->18 (exact)
    const int NSUB  = 8 / MT;             // N subtile groups across waves
    const int NBLK  = NSUB * NACC * 16;   // positions per block
    const int CSH   = (CIN == 32) ? 5 : 6;
    const int AELEM = MT * nKb * 512;     // f16 elements of A in LDS

    __shared__ _Float16 ldsA[AELEM];
    __shared__ _Float16 ldsB[NBLK][32];

    int b      = blockIdx.y;
    int mtile0 = blockIdx.z * MT;
    int pbase  = blockIdx.x * NBLK;
    int tid    = threadIdx.x;
    int wv     = tid >> 5, lane = tid & 31;
    int mtl    = wv % MT, nsub = wv / MT;

    // ---- TDM: DMA this block's A pack (MT rows of nKb*512 f16) into LDS ----
    if (wv == 0) {
        const int nMtTot = COUT / 16;
        unsigned long long ga = (unsigned long long)(size_t)
            (wpack + (((size_t)b * nMtTot + mtile0) * nKb) * 512);
        unsigned int ldsaddr = (unsigned int)(size_t)(&ldsA[0]); // flat addr low 32 = LDS offset
        const unsigned int rows = (unsigned int)(MT * nKb);      // 36 rows of 512 f16
        u32x4 g0;
        g0[0] = 1u;                                   // count=1, user descriptor
        g0[1] = ldsaddr;                              // lds_addr
        g0[2] = (unsigned int)(ga & 0xFFFFFFFFull);   // global_addr[31:0]
        g0[3] = (unsigned int)((ga >> 32) & 0x1FFFFFFull) | (2u << 30); // ga[56:32], type=2
        i32x8 g1;
        g1[0] = (int)(1u << 16);                      // workgroup_mask=0, data_size=1 (2B)
        g1[1] = (int)(512u << 16);                    // tensor_dim0[15:0]=512
        g1[2] = (int)(rows << 16);                    // tensor_dim0 hi=0, tensor_dim1=rows
        g1[3] = (int)(512u << 16);                    // tensor_dim1 hi=0, tile_dim0=512
        g1[4] = (int)rows;                            // tile_dim1=rows, tile_dim2=0
        g1[5] = 512;                                  // tensor_dim0_stride=512
        g1[6] = 0; g1[7] = 0;
        i32x4 gz4 = {0, 0, 0, 0};
        i32x8 gz8 = {0, 0, 0, 0, 0, 0, 0, 0};
        __builtin_amdgcn_tensor_load_to_lds(g0, g1, gz4, gz4, gz8, 0);
        __builtin_amdgcn_s_wait_tensorcnt((short)0);
    }
    __syncthreads();

    const _Float16* inb = hin + (size_t)b * CIN * HIN * WIN;

    v8f acc[NACC];
    for (int a = 0; a < NACC; ++a) acc[a] = v8f{};

    for (int kblk = 0; kblk < nKb; ++kblk) {
        // cooperative im2col fill (position-major for coalescing along W)
        for (int i = tid; i < NBLK * 32; i += 256) {
            int n = i % NBLK, k = i / NBLK;
            int p = pbase + n;
            int y = p / WOUT, xo = p % WOUT;
            int K = kblk * 32 + k;
            int c  = K & (CIN - 1);          // tap-major K
            int t9 = K >> CSH;
            int ky = t9 / 3, kx = t9 % 3;
            int hi = y * STRIDE - 1 + ky, wi = xo * STRIDE - 1 + kx;
            _Float16 val = (_Float16)0.f;
            if (hi >= 0 && hi < HIN && wi >= 0 && wi < WIN)
                val = inb[(size_t)c * HIN * WIN + (size_t)hi * WIN + wi];
            ldsB[n][k] = val;
        }
        __syncthreads();

        v16h afrag = *reinterpret_cast<const v16h*>(
            &ldsA[((size_t)mtl * nKb + kblk) * 512 + lane * 16]);
        for (int a = 0; a < NACC; ++a) {
            v16h bfrag = *reinterpret_cast<const v16h*>(
                &ldsB[(nsub * NACC + a) * 16 + (lane & 15)][(lane < 16) ? 0 : 16]);
            acc[a] = __builtin_amdgcn_wmma_f32_16x16x32_f16(false, afrag, false, bfrag,
                                                            (short)0, acc[a], false, false);
        }
        __syncthreads();
    }

    // epilogue: C/D layout -> VGPR r holds M = r (lanes 0-15) / r+8 (lanes 16-31)
    int mofs = (lane < 16) ? 0 : 8;
    for (int a = 0; a < NACC; ++a) {
        int n = (nsub * NACC + a) * 16 + (lane & 15);
        int p = pbase + n;
        for (int rr = 0; rr < 8; ++rr) {
            int cout = (mtile0 + mtl) * 16 + rr + mofs;
            float sc  = g[cout] * rsqrtf(vv[cout] + BN_EPS);
            float val = (acc[a][rr] + cb[b * COUT + cout]) * sc + (be[cout] - m[cout] * sc);
            hout[((size_t)b * COUT + cout) * (HOUT * WOUT) + p] = (_Float16)fmaxf(val, 0.f);
        }
    }
}

// ---------------------------------------------------------------------------
// Head: out[b][j] = pooled3[b] . fc_w[j] + fc_b[j]   (64x2, fp32)
// ---------------------------------------------------------------------------
__global__ void head_kernel(const float* __restrict__ pooled, const float* __restrict__ fw,
                            const float* __restrict__ fb, float* __restrict__ out)
{
    int t = threadIdx.x;
    if (t >= 128) return;
    int b = t >> 1, j = t & 1;
    float s = fb[j];
    for (int c = 0; c < 128; ++c) s += pooled[b * 128 + c] * fw[j * 128 + c];
    out[t] = s;
}

// ---------------------------------------------------------------------------
extern "C" void kernel_launch(void* const* d_in, const int* in_sizes, int n_in,
                              void* d_out, int out_size, void* d_ws, size_t ws_size,
                              hipStream_t stream) {
    const float* x      = (const float*)d_in[0];
    const float* stem_w = (const float*)d_in[1];
    const float* bn0_g  = (const float*)d_in[2];
    const float* bn0_b  = (const float*)d_in[3];
    const float* bn0_m  = (const float*)d_in[4];
    const float* bn0_v  = (const float*)d_in[5];
    const float* r1_w   = (const float*)d_in[6];
    const float* r1_b   = (const float*)d_in[7];
    const float* c1_w   = (const float*)d_in[8];
    const float* c1_b   = (const float*)d_in[9];
    const float* bn1_g  = (const float*)d_in[10];
    const float* bn1_b  = (const float*)d_in[11];
    const float* bn1_m  = (const float*)d_in[12];
    const float* bn1_v  = (const float*)d_in[13];
    const float* r2_w   = (const float*)d_in[14];
    const float* r2_b   = (const float*)d_in[15];
    const float* c2_w   = (const float*)d_in[16];
    const float* c2_b   = (const float*)d_in[17];
    const float* bn2_g  = (const float*)d_in[18];
    const float* bn2_b  = (const float*)d_in[19];
    const float* bn2_m  = (const float*)d_in[20];
    const float* bn2_v  = (const float*)d_in[21];
    const float* fc_w   = (const float*)d_in[22];
    const float* fc_b   = (const float*)d_in[23];
    float* out = (float*)d_out;

    // workspace carve-up
    size_t off = 0;
    char* ws = (char*)d_ws;
    auto carve = [&](size_t bytes) -> void* {
        void* p = ws + off; off += (bytes + 255) & ~(size_t)255; return p;
    };
    const size_t S1 = 112 * 112, S2 = 56 * 56;
    _Float16* h1   = (_Float16*)carve((size_t)64 * 32 * S1 * 2);
    _Float16* h2   = (_Float16*)carve((size_t)64 * 64 * S1 * 2);
    _Float16* h3   = (_Float16*)carve((size_t)64 * 128 * S2 * 2);
    float* pooled1 = (float*)carve(64 * 32 * 4);
    float* rt1     = (float*)carve(64 * 4 * 4);
    _Float16* cw1  = (_Float16*)carve((size_t)64 * 4 * 9 * 512 * 2);
    float* cb1     = (float*)carve(64 * 64 * 4);
    float* pooled2 = (float*)carve(64 * 64 * 4);
    float* rt2     = (float*)carve(64 * 4 * 4);
    _Float16* cw2  = (_Float16*)carve((size_t)64 * 8 * 18 * 512 * 2);
    float* cb2     = (float*)carve(64 * 128 * 4);
    float* pooled3 = (float*)carve(64 * 128 * 4);

    // 1) stem conv + BN + ReLU
    stem_kernel<<<(64 * 32 * (int)S1 + 255) / 256, 256, 0, stream>>>(
        x, stem_w, bn0_g, bn0_b, bn0_m, bn0_v, h1);

    // 2) route 1
    gap_kernel<112 * 112><<<64 * 32, 256, 0, stream>>>(h1, pooled1);
    route_kernel<32><<<1, 256, 0, stream>>>(pooled1, r1_w, r1_b, rt1);
    mixw_kernel<32, 64><<<(64 * 4 * 9 * 32) / 256, 256, 0, stream>>>(rt1, c1_w, cw1);
    mixb_kernel<64><<<(64 * 64) / 256, 256, 0, stream>>>(rt1, c1_b, cb1);

    // 3) condconv1 (32->64, s1 p1): MT=4 (all Cout), NACC=2 -> 64 pos/block [WMMA+TDM]
    condconv_wmma<32, 64, 4, 2, 1, 112, 112, 112, 112>
        <<<dim3(112 * 112 / 64, 64, 1), 256, 0, stream>>>(
            h1, cw1, cb1, bn1_g, bn1_b, bn1_m, bn1_v, h2);

    // 4) route 2
    gap_kernel<112 * 112><<<64 * 64, 256, 0, stream>>>(h2, pooled2);
    route_kernel<64><<<1, 256, 0, stream>>>(pooled2, r2_w, r2_b, rt2);
    mixw_kernel<64, 128><<<(64 * 8 * 18 * 32) / 256, 256, 0, stream>>>(rt2, c2_w, cw2);
    mixb_kernel<128><<<(64 * 128) / 256, 256, 0, stream>>>(rt2, c2_b, cb2);

    // 5) condconv2 (64->128, s2 p1): MT=2 per block (grid.z=4), NACC=1 [WMMA+TDM]
    condconv_wmma<64, 128, 2, 1, 2, 112, 112, 56, 56>
        <<<dim3(56 * 56 / 64, 64, 4), 256, 0, stream>>>(
            h2, cw2, cb2, bn2_g, bn2_b, bn2_m, bn2_v, h3);

    // 6) head: GAP -> linear
    gap_kernel<56 * 56><<<64 * 128, 256, 0, stream>>>(h3, pooled3);
    head_kernel<<<1, 128, 0, stream>>>(pooled3, fc_w, fc_b, out);
}
